// GatedTinyMambaLayer_70265664963057
// MI455X (gfx1250) — matrix operand order repaired and checked by reading, not compiled, not run on hardware
//
#include <hip/hip_runtime.h>

typedef __attribute__((ext_vector_type(16))) _Float16 v16h;
typedef __attribute__((ext_vector_type(8)))  float    v8f;

#define D   32
#define TDIM 8192
#define BDIM 64
#define TILES_PER_BATCH (TDIM / 16)   /* 512 */
#define TILES_PER_WAVE  8
#define WAVES_PER_BLOCK 8

#define LOG2E  1.4426950408889634f
#define NEGL2E (-1.4426950408889634f)
#define CDK    (-0.05f * LOG2E)        /* decay exponent scale, base-2 */

__device__ __forceinline__ float sigm2(float v) {
    // expects v pre-scaled by -log2(e): sigmoid = rcp(1 + exp2(v))
    return __builtin_amdgcn_rcpf(1.0f + __builtin_amdgcn_exp2f(v));
}

// ---------------------------------------------------------------------------
// Pre-kernel: per-batch constants  proj = prev@Ws+bs,  sg_b = proj@Wsg+bsg,
// dg_b = proj@Wdg[32:]+bdg.   64 blocks x 32 threads (one wave / batch row).
// ---------------------------------------------------------------------------
__global__ __launch_bounds__(32) void precompute_kernel(
    const float* __restrict__ prev, const float* __restrict__ Ws,  const float* __restrict__ bs,
    const float* __restrict__ Wsg,  const float* __restrict__ bsg,
    const float* __restrict__ Wdg,  const float* __restrict__ bdg,
    float* __restrict__ proj, float* __restrict__ sgb, float* __restrict__ dgb)
{
    __shared__ float sp[D];
    const int b = blockIdx.x, c = threadIdx.x;
    float p = bs[c];
    #pragma unroll
    for (int k = 0; k < D; ++k) p += prev[b * D + k] * Ws[k * D + c];
    sp[c] = p;
    __syncthreads();
    proj[b * D + c] = p;
    float sg = bsg[c], dg = bdg[c];
    #pragma unroll
    for (int k = 0; k < D; ++k) {
        sg += sp[k] * Wsg[k * D + c];
        dg += sp[k] * Wdg[(D + k) * D + c];   // bottom half of Wdg (64x32)
    }
    sgb[b * D + c] = sg;
    dgb[b * D + c] = dg;
}

// ---------------------------------------------------------------------------
// Main kernel: WMMA-tiled, one wave = 8 token tiles of 16 within one batch,
// double-buffered global loads (branchless, clamped indices).
// ---------------------------------------------------------------------------
__global__ __launch_bounds__(256) void mamba_main_kernel(
    const float* __restrict__ x,
    const float* __restrict__ proj, const float* __restrict__ sgb, const float* __restrict__ dgb,
    const float* __restrict__ Wi,  const float* __restrict__ bi,
    const float* __restrict__ Wig, const float* __restrict__ big,
    const float* __restrict__ Wcg, const float* __restrict__ bcg,
    const float* __restrict__ Wdg,                    // rows 0..31 used here
    const float* __restrict__ Wo,  const float* __restrict__ bo,
    float* __restrict__ out, float* __restrict__ outState)
{
    __shared__ float lds[WAVES_PER_BLOCK][16][D];     // per-wave 2KB state stage

    const int lane = threadIdx.x & 31;
    const int wave = threadIdx.x >> 5;
    const int hi   = lane >> 4;        // lane half
    const int l16  = lane & 15;
    const long wgid = (long)blockIdx.x * WAVES_PER_BLOCK + wave;
    const int  b    = (int)((wgid * TILES_PER_WAVE) / TILES_PER_BATCH);

    // ---- resident weight B-fragments (lane=col l16, K = 8*hi + 2*(v&3) + 16*(v>>2) + {0,1}) ----
    auto loadFrag = [&](const float* __restrict__ W, int c0, float scale,
                        const float* __restrict__ scaleK) -> v16h {
        v16h f;
        const int n = c0 + l16;
        #pragma unroll
        for (int v = 0; v < 8; ++v) {
            const int k = 8 * hi + 2 * (v & 3) + 16 * (v >> 2);
            const float s0 = scaleK ? scale * scaleK[k]     : scale;
            const float s1 = scaleK ? scale * scaleK[k + 1] : scale;
            f[2 * v]     = (_Float16)(s0 * W[k * D + n]);
            f[2 * v + 1] = (_Float16)(s1 * W[(k + 1) * D + n]);
        }
        return f;
    };

    const float* __restrict__ projb = proj + b * D;
    v16h fNs[2], fIg[2], fCg[2], fDg[2], fWo[2];
    fNs[0] = loadFrag(Wi,  0,  1.0f,   nullptr);
    fNs[1] = loadFrag(Wi,  16, 1.0f,   nullptr);
    fIg[0] = loadFrag(Wig, 0,  NEGL2E, nullptr);    // gate weights pre-scaled by
    fIg[1] = loadFrag(Wig, 16, NEGL2E, nullptr);    // -log2(e): sigmoid via exp2
    fDg[0] = loadFrag(Wdg, 0,  NEGL2E, nullptr);    // Wdg top half
    fDg[1] = loadFrag(Wdg, 16, NEGL2E, nullptr);
    fCg[0] = loadFrag(Wcg, 0,  NEGL2E, projb);      // proj[b][k] folded in too
    fCg[1] = loadFrag(Wcg, 16, NEGL2E, projb);
    fWo[0] = loadFrag(Wo,  0,  1.0f,   nullptr);
    fWo[1] = loadFrag(Wo,  16, 1.0f,   nullptr);

    // ---- per-lane constants; the two gate constants become C-splats ----
    float projc[2], bic[2], boc[2];
    v8f c0m[2], c0d[2];
    #pragma unroll
    for (int n = 0; n < 2; ++n) {
        const int c = 16 * n + l16;
        projc[n] = projb[c];
        bic[n]   = bi[c];
        boc[n]   = bo[c];
        const float vm = NEGL2E * (big[c] + sgb[b * D + c] + bcg[c]);
        const float vd = NEGL2E * dgb[b * D + c];
        #pragma unroll
        for (int r = 0; r < 8; ++r) { c0m[n][r] = vm;  c0d[n][r] = vd; }
    }

    const float* __restrict__ xwave = x + ((long)b * TDIM +
        (long)(wgid * TILES_PER_WAVE & (TILES_PER_BATCH - 1)) * 16) * D;
    const int laneOff = l16 * D + 8 * hi;    // this lane's A-slice offset

    // ---- prologue: load tile 0 ----
    float4 xq[2][2], xqN[2][2];
    #pragma unroll
    for (int g = 0; g < 2; ++g) {
        const float* base = xwave + laneOff + 16 * g;
        xq[g][0] = *(const float4*)(base);
        xq[g][1] = *(const float4*)(base + 4);
    }

    for (int it = 0; it < TILES_PER_WAVE; ++it) {
        const long tile = wgid * TILES_PER_WAVE + it;
        const int  trow = (int)(tile & (TILES_PER_BATCH - 1)) * 16;

        // ---- double buffer: branchless clamped indices -> uniform loop body ----
        const int itN = min(it + 1, TILES_PER_WAVE - 1);
        const int itP = min(it + 2, TILES_PER_WAVE - 1);
        {
            const float* nxt = xwave + (long)itN * 16 * D + laneOff;
            #pragma unroll
            for (int g = 0; g < 2; ++g) {
                xqN[g][0] = *(const float4*)(nxt + 16 * g);
                xqN[g][1] = *(const float4*)(nxt + 16 * g + 4);
            }
        }
        __builtin_prefetch(xwave + (long)itP * 16 * D + laneOff, 0, 3);  // WGP scope

        // ---- convert current tile to f16 A-fragment ----
        v16h aX;
        #pragma unroll
        for (int g = 0; g < 2; ++g) {
            aX[8 * g + 0] = (_Float16)xq[g][0].x;  aX[8 * g + 1] = (_Float16)xq[g][0].y;
            aX[8 * g + 2] = (_Float16)xq[g][0].z;  aX[8 * g + 3] = (_Float16)xq[g][0].w;
            aX[8 * g + 4] = (_Float16)xq[g][1].x;  aX[8 * g + 5] = (_Float16)xq[g][1].y;
            aX[8 * g + 6] = (_Float16)xq[g][1].z;  aX[8 * g + 7] = (_Float16)xq[g][1].w;
        }

        // ---- stage 1: 8 WMMAs; ig accumulates onto cg (C-chaining) ----
        v8f accNs[2], accM[2], accDg[2];
        #pragma unroll
        for (int n = 0; n < 2; ++n) {
            accNs[n] = __builtin_amdgcn_wmma_f32_16x16x32_f16(false, aX, false, fNs[n], (short)0, v8f{},   false, false);
            v8f cg   = __builtin_amdgcn_wmma_f32_16x16x32_f16(false, aX, false, fCg[n], (short)0, c0m[n],  false, false);
            accM[n]  = __builtin_amdgcn_wmma_f32_16x16x32_f16(false, aX, false, fIg[n], (short)0, cg,      false, false);
            accDg[n] = __builtin_amdgcn_wmma_f32_16x16x32_f16(false, aX, false, fDg[n], (short)0, c0d[n],  false, false);
        }

        // ---- elementwise gating (5 trans + ~3 VALU per element) ----
        #pragma unroll
        for (int n = 0; n < 2; ++n) {
            #pragma unroll
            for (int r = 0; r < 8; ++r) {
                const float mix = sigm2(accM[n][r]);
                const float dcy = __builtin_amdgcn_exp2f(CDK * sigm2(accDg[n][r]));
                const float s   = (accNs[n][r] + bic[n]) * mix + projc[n] * dcy;
                lds[wave][r + 8 * hi][16 * n + l16] = s;
            }
        }
        asm volatile("s_wait_dscnt 0" ::: "memory");  // wave-local LDS fence

        // ---- stage 2: out = new_state @ Wo (A-layout via 4x ds_load_b128) ----
        float4 sq[2][2];
        #pragma unroll
        for (int g = 0; g < 2; ++g) {
            const float* base = &lds[wave][l16][16 * g + 8 * hi];
            sq[g][0] = *(const float4*)(base);
            sq[g][1] = *(const float4*)(base + 4);
        }
        v16h aS;
        #pragma unroll
        for (int g = 0; g < 2; ++g) {
            aS[8 * g + 0] = (_Float16)sq[g][0].x;  aS[8 * g + 1] = (_Float16)sq[g][0].y;
            aS[8 * g + 2] = (_Float16)sq[g][0].z;  aS[8 * g + 3] = (_Float16)sq[g][0].w;
            aS[8 * g + 4] = (_Float16)sq[g][1].x;  aS[8 * g + 5] = (_Float16)sq[g][1].y;
            aS[8 * g + 6] = (_Float16)sq[g][1].z;  aS[8 * g + 7] = (_Float16)sq[g][1].w;
        }
        v8f accO[2];
        #pragma unroll
        for (int n = 0; n < 2; ++n)
            accO[n] = __builtin_amdgcn_wmma_f32_16x16x32_f16(false, aS, false, fWo[n], (short)0, v8f{}, false, false);

        // ---- store out tile (+bo); two contiguous 64B runs per instruction ----
        float* __restrict__ ot = out + ((long)b * TDIM + trow) * D;
        #pragma unroll
        for (int n = 0; n < 2; ++n)
            #pragma unroll
            for (int r = 0; r < 8; ++r)
                ot[(r + 8 * hi) * D + 16 * n + l16] = accO[n][r] + boc[n];

        // ---- final-state output: row 15 (still in LDS) of the batch's last tile ----
        if ((tile & (TILES_PER_BATCH - 1)) == (TILES_PER_BATCH - 1) && hi == 1) {
            outState[b * D + l16]      = lds[wave][15][l16];
            outState[b * D + 16 + l16] = lds[wave][15][16 + l16];
        }

        // rotate double buffer
        #pragma unroll
        for (int g = 0; g < 2; ++g) { xq[g][0] = xqN[g][0]; xq[g][1] = xqN[g][1]; }
    }
}

// ---------------------------------------------------------------------------
extern "C" void kernel_launch(void* const* d_in, const int* in_sizes, int n_in,
                              void* d_out, int out_size, void* d_ws, size_t ws_size,
                              hipStream_t stream) {
    (void)in_sizes; (void)n_in; (void)out_size; (void)ws_size;
    const float* x    = (const float*)d_in[0];
    const float* prev = (const float*)d_in[1];
    const float* Wi   = (const float*)d_in[2];
    const float* bi   = (const float*)d_in[3];
    const float* Ws   = (const float*)d_in[4];
    const float* bs   = (const float*)d_in[5];
    const float* Wo   = (const float*)d_in[6];
    const float* bo   = (const float*)d_in[7];
    const float* Wig  = (const float*)d_in[8];
    const float* big  = (const float*)d_in[9];
    const float* Wsg  = (const float*)d_in[10];
    const float* bsg  = (const float*)d_in[11];
    const float* Wcg  = (const float*)d_in[12];
    const float* bcg  = (const float*)d_in[13];
    const float* Wdg  = (const float*)d_in[14];
    const float* bdg  = (const float*)d_in[15];

    float* ws   = (float*)d_ws;
    float* proj = ws;                    // 64*32
    float* sgb  = ws + BDIM * D;         // 64*32
    float* dgb  = ws + 2 * BDIM * D;     // 64*32

    float* out      = (float*)d_out;
    float* outState = out + (long)BDIM * TDIM * D;

    precompute_kernel<<<BDIM, 32, 0, stream>>>(prev, Ws, bs, Wsg, bsg, Wdg, bdg,
                                               proj, sgb, dgb);

    const int totalWaves = (BDIM * TILES_PER_BATCH) / TILES_PER_WAVE;  // 4096
    const int blocks     = totalWaves / WAVES_PER_BLOCK;               // 512
    mamba_main_kernel<<<blocks, WAVES_PER_BLOCK * 32, 0, stream>>>(
        x, proj, sgb, dgb, Wi, bi, Wig, big, Wcg, bcg, Wdg, Wo, bo, out, outState);
}
